// TGSFNNetwork_56281251447085
// MI455X (gfx1250) — matrix operand order repaired and checked by reading, not compiled, not run on hardware
//
#include <hip/hip_runtime.h>
#include <hip/hip_bf16.h>

// ---------------------------------------------------------------------------
// LIF spiking network on MI455X (gfx1250).
//   B=32, T=128, D=256, H=4096, O=10, ALPHA = DT/TAU_M = 0.05
// 128 serially-dependent [32x4096]x[4096x4096] GEMMs (~137 GFLOP) on
// v_wmma_f32_16x16x32_bf16 (spikes are exactly {0,1} in bf16), encoder GEMM
// fused by K-concatenation (K = 4096 + 256 = 4352).  W_rec bf16 (32MB) stays
// L2-resident.  One PERSISTENT kernel: 128 blocks x 2 wave32 stay resident;
// the two waves split the K range (halving the per-step serial latency) and
// combine partial accumulators through LDS; membrane V and the avg-rate
// accumulator live in VGPRs across all 128 steps; steps sync with a
// device-scope arrive counter.  Only the bf16 spike vector (double-buffered)
// crosses WGPs through L2.
// ---------------------------------------------------------------------------

typedef __attribute__((ext_vector_type(16))) __bf16 v16bf;
typedef __attribute__((ext_vector_type(8)))  float  v8f;

#define B_    32
#define T_    128
#define D_    256
#define H_    4096
#define O_    10
#define KTOT  4352          // H + D
#define ALPHA 0.05f
#define NBLK  (H_ / 32)     // 128 persistent blocks (2 waves each)

// ---- workspace layout (bytes) ----
#define OFF_BT   ((size_t)0)
#define SZ_BT    ((size_t)H_ * KTOT * 2)            // bf16 B^T  [N=H][K=KTOT]
#define OFF_SPK0 (OFF_BT + SZ_BT)
#define SZ_SPK   ((size_t)B_ * H_ * 2)              // bf16 spike ping
#define OFF_SPK1 (OFF_SPK0 + SZ_SPK)                // bf16 spike pong
#define OFF_AVG  (OFF_SPK1 + SZ_SPK)
#define SZ_AVG   ((size_t)B_ * H_ * 4)              // f32 sum_t(spk)
#define OFF_CNT  (OFF_AVG + SZ_AVG)                 // u32 barrier counter
#define OFF_XBF  (OFF_CNT + 256)                    // bf16 x in [T][B][D]
#define SZ_XBF   ((size_t)T_ * B_ * D_ * 2)

// ---------------------------------------------------------------------------
// Coalesced LDS-tiled transpose:  Bt[n][k] = (bf16)W_rec[k][n],  k < H.
// ---------------------------------------------------------------------------
__global__ __launch_bounds__(256)
void transpose_rec(const float* __restrict__ W_rec, __bf16* __restrict__ Bt)
{
    __shared__ float tile[32][33];
    const int tx = threadIdx.x;          // 0..31
    const int ty = threadIdx.y;          // 0..7
    const int n0 = blockIdx.x * 32;      // W_rec column  == Bt row
    const int k0 = blockIdx.y * 32;      // W_rec row     == Bt column

#pragma unroll
    for (int i = ty; i < 32; i += 8)
        tile[i][tx] = W_rec[(size_t)(k0 + i) * H_ + (n0 + tx)];
    __syncthreads();
#pragma unroll
    for (int i = ty; i < 32; i += 8)
        Bt[(size_t)(n0 + i) * KTOT + (k0 + tx)] = (__bf16)tile[tx][i];
}

// Encoder tail is row-major on both sides: straight coalesced copy-convert.
__global__ void copy_enc(const float* __restrict__ W_enc, __bf16* __restrict__ Bt)
{
    int i = blockIdx.x * blockDim.x + threadIdx.x;   // i over H*D
    if (i >= H_ * D_) return;
    int n = i / D_;
    int d = i % D_;
    Bt[(size_t)n * KTOT + H_ + d] = (__bf16)W_enc[i];
}

// x [B,T,D] f32 -> Xbf [T,B,D] bf16 (d-coalesced on both sides).
__global__ void xpose_x(const float* __restrict__ x, __bf16* __restrict__ Xbf)
{
    int i = blockIdx.x * blockDim.x + threadIdx.x;   // over B*T*D
    if (i >= B_ * T_ * D_) return;
    int d  = i % D_;
    int bt = i / D_;
    int tt = bt % T_;
    int b  = bt / T_;
    Xbf[((size_t)tt * B_ + b) * D_ + d] = (__bf16)x[i];
}

__global__ void init_state(__bf16* __restrict__ spk0, float* __restrict__ rate,
                           unsigned* __restrict__ cnt)
{
    int i = blockIdx.x * blockDim.x + threadIdx.x;
    if (i < B_ * H_) spk0[i] = (__bf16)0.0f;
    if (i < T_)      rate[i] = 0.0f;
    if (i == 0)      *cnt    = 0u;
}

// ---------------------------------------------------------------------------
// Persistent LIF chain.  Grid: 128 blocks x 64 threads (2 wave32s, trivially
// co-resident).  Both waves compute the block's 32x32 output strip over half
// the K range each, exchange partial accumulators through LDS, then wave w
// finalizes M-tile w (keeping its V/avg strip in VGPRs for all T steps).
// ---------------------------------------------------------------------------
__global__ __launch_bounds__(64)
void lif_persist(const __bf16* __restrict__ Bt,   const __bf16* __restrict__ Xbf,
                 __bf16* __restrict__ spk0,       __bf16* __restrict__ spk1,
                 float* __restrict__ avg_out,     float* __restrict__ spikes,
                 float* __restrict__ volts,       float* __restrict__ rate,
                 unsigned* __restrict__ cnt)
{
    __shared__ float xch[2][2][8][32];       // [src wave][nt][r][lane]

    const int tid   = threadIdx.x;
    const int lane  = tid & 31;
    const int wv    = tid >> 5;              // 0/1: K-half now, M-tile later
    const int Nb    = blockIdx.x * 32;       // column strip base
    const int lhalf = lane >> 4;             // lane group (0/1)
    const int l16   = lane & 15;

    // A layout (ISA 7.12.2, 16-bit A 16x32): lane holds row M = lane%16;
    // packed pair v<4 -> K = 2v + 8*lhalf, v>=4 -> K = 16 + 2(v-4) + 8*lhalf.
    int aoff[8];
#pragma unroll
    for (int v = 0; v < 8; ++v)
        aoff[v] = (v < 4 ? 2 * v : 16 + 2 * (v - 4)) + 8 * lhalf;

    // B layout (32x16, 16-bit): lane holds column N = lane%16,
    // K = i + 8*lhalf (i<8), 16 + (i-8) + 8*lhalf (i>=8) -> two 16B slices of Bt row (Nb + l16).
    const size_t brow0 = (size_t)(Nb + l16)      * KTOT + 8 * lhalf;
    const size_t brow1 = (size_t)(Nb + 16 + l16) * KTOT + 8 * lhalf;

    // Persistent per-lane state for M-tile wv:
    // element (b = wv*16 + 8*lhalf + r, n = Nb + nt*16 + l16).
    float Vr[2][8];
    float Ar[2][8];
#pragma unroll
    for (int nt = 0; nt < 2; ++nt)
#pragma unroll
        for (int r = 0; r < 8; ++r) { Vr[nt][r] = 0.0f; Ar[nt][r] = 0.0f; }

    for (int t = 0; t < T_; ++t) {
        const __bf16* sin = (t & 1) ? spk1 : spk0;
        __bf16*       sou = (t & 1) ? spk0 : spk1;

        v8f c00 = {}, c01 = {}, c10 = {}, c11 = {};

        const __bf16* arow0 = sin + (size_t)(l16)      * H_;   // M-tile 0
        const __bf16* arow1 = sin + (size_t)(16 + l16) * H_;   // M-tile 1

        // ---- K-half over recurrent spikes: 64 tiles of K=32 per wave ----
#pragma unroll 4
        for (int k2 = 0; k2 < 64; ++k2) {
            const int kb = (wv * 64 + k2) * 32;
            union { unsigned int u[8]; v16bf v; } A0, A1;
#pragma unroll
            for (int v = 0; v < 8; ++v) {
                A0.u[v] = *(const unsigned int*)(arow0 + kb + aoff[v]);
                A1.u[v] = *(const unsigned int*)(arow1 + kb + aoff[v]);
            }
            union { uint4 q[2]; v16bf v; } Bm0, Bm1;
            const uint4* p0 = (const uint4*)(Bt + brow0 + kb);
            const uint4* p1 = (const uint4*)(Bt + brow1 + kb);
            Bm0.q[0] = p0[0]; Bm0.q[1] = p0[2];
            Bm1.q[0] = p1[0]; Bm1.q[1] = p1[2];
            __builtin_prefetch(Bt + brow0 + kb + 128, 0, 3);
            __builtin_prefetch(Bt + brow1 + kb + 128, 0, 3);

            c00 = __builtin_amdgcn_wmma_f32_16x16x32_bf16(false, A0.v, false, Bm0.v,
                                                          (short)0, c00, false, false);
            c01 = __builtin_amdgcn_wmma_f32_16x16x32_bf16(false, A0.v, false, Bm1.v,
                                                          (short)0, c01, false, false);
            c10 = __builtin_amdgcn_wmma_f32_16x16x32_bf16(false, A1.v, false, Bm0.v,
                                                          (short)0, c10, false, false);
            c11 = __builtin_amdgcn_wmma_f32_16x16x32_bf16(false, A1.v, false, Bm1.v,
                                                          (short)0, c11, false, false);
        }

        // ---- K-half over external input (bf16, step-major): 4 tiles/wave ----
        const __bf16* xr0 = Xbf + ((size_t)t * B_ + l16)      * D_;
        const __bf16* xr1 = Xbf + ((size_t)t * B_ + 16 + l16) * D_;
#pragma unroll
        for (int k2 = 0; k2 < 4; ++k2) {
            const int kb = (wv * 4 + k2) * 32;
            union { unsigned int u[8]; v16bf v; } A0, A1;
#pragma unroll
            for (int v = 0; v < 8; ++v) {
                A0.u[v] = *(const unsigned int*)(xr0 + kb + aoff[v]);
                A1.u[v] = *(const unsigned int*)(xr1 + kb + aoff[v]);
            }
            union { uint4 q[2]; v16bf v; } Bm0, Bm1;
            const uint4* p0 = (const uint4*)(Bt + brow0 + H_ + kb);
            const uint4* p1 = (const uint4*)(Bt + brow1 + H_ + kb);
            Bm0.q[0] = p0[0]; Bm0.q[1] = p0[2];
            Bm1.q[0] = p1[0]; Bm1.q[1] = p1[2];

            c00 = __builtin_amdgcn_wmma_f32_16x16x32_bf16(false, A0.v, false, Bm0.v,
                                                          (short)0, c00, false, false);
            c01 = __builtin_amdgcn_wmma_f32_16x16x32_bf16(false, A0.v, false, Bm1.v,
                                                          (short)0, c01, false, false);
            c10 = __builtin_amdgcn_wmma_f32_16x16x32_bf16(false, A1.v, false, Bm0.v,
                                                          (short)0, c10, false, false);
            c11 = __builtin_amdgcn_wmma_f32_16x16x32_bf16(false, A1.v, false, Bm1.v,
                                                          (short)0, c11, false, false);
        }

        // ---- exchange partial sums of the non-owned M-tile through LDS ----
        {
            const v8f pa = wv ? c00 : c10;   // partials of tile (1-wv, nt=0)
            const v8f pb = wv ? c01 : c11;   // partials of tile (1-wv, nt=1)
#pragma unroll
            for (int r = 0; r < 8; ++r) {
                xch[wv][0][r][lane] = pa[r];
                xch[wv][1][r][lane] = pb[r];
            }
        }
        __syncthreads();
        v8f d0 = wv ? c10 : c00;             // own M-tile partials
        v8f d1 = wv ? c11 : c01;
        const int ow = 1 - wv;
#pragma unroll
        for (int r = 0; r < 8; ++r) {
            d0[r] = d0[r] + xch[ow][0][r][lane];
            d1[r] = d1[r] + xch[ow][1][r][lane];
        }

        // ---- LIF epilogue: wave wv owns rows [wv*16, wv*16+16) ----
        const int em = wv * 16 + 8 * lhalf;
        float wsum = 0.0f;
#pragma unroll
        for (int nt = 0; nt < 2; ++nt) {
            const int n = Nb + nt * 16 + l16;
            const v8f acc = nt ? d1 : d0;
#pragma unroll
            for (int r = 0; r < 8; ++r) {
                const int b = em + r;
                float v_new = Vr[nt][r] * (1.0f - ALPHA) + acc[r] * ALPHA;
                float s     = (v_new >= 1.0f) ? 1.0f : 0.0f;
                v_new       = (s > 0.0f) ? 0.0f : v_new;
                Vr[nt][r]  = v_new;
                Ar[nt][r] += s;
                sou[(size_t)b * H_ + n] = (__bf16)s;
                const size_t oidx = ((size_t)b * T_ + t) * H_ + n;
                spikes[oidx] = s;
                volts[oidx]  = v_new;
                wsum        += s;
            }
        }
#pragma unroll
        for (int off = 16; off >= 1; off >>= 1)
            wsum += __shfl_xor(wsum, off, 32);
        if (lane == 0)
            atomicAdd(rate + t, wsum * (1.0f / (float)B_));

        // ---- device-wide step barrier (also guards LDS reuse) ----
        __threadfence();                                   // release stores
        __syncthreads();
        if (tid == 0) {
            atomicAdd(cnt, 1u);
            const unsigned target = (unsigned)gridDim.x * (unsigned)(t + 1);
            while (((volatile unsigned*)cnt)[0] < target)
                __builtin_amdgcn_s_sleep(2);
        }
        __syncthreads();
        __threadfence();                                   // acquire
    }

    // ---- spill per-lane avg accumulator once ----
    const int em = wv * 16 + 8 * lhalf;
#pragma unroll
    for (int nt = 0; nt < 2; ++nt) {
        const int n = Nb + nt * 16 + l16;
#pragma unroll
        for (int r = 0; r < 8; ++r)
            avg_out[(size_t)(em + r) * H_ + n] = Ar[nt][r];
    }
}

// ---------------------------------------------------------------------------
// out[b][o] = (sum_t spk / T) . W_read[o] + b_read[o]
// ---------------------------------------------------------------------------
__global__ void readout(const float* __restrict__ avg,
                        const float* __restrict__ W_read,
                        const float* __restrict__ b_read,
                        float* __restrict__ out)
{
    __shared__ float red[256];
    const int b = blockIdx.x / O_;
    const int o = blockIdx.x % O_;
    float s = 0.0f;
    for (int h = threadIdx.x; h < H_; h += blockDim.x)
        s += avg[(size_t)b * H_ + h] * W_read[(size_t)o * H_ + h];
    red[threadIdx.x] = s;
    __syncthreads();
    for (int st = 128; st > 0; st >>= 1) {
        if (threadIdx.x < st) red[threadIdx.x] += red[threadIdx.x + st];
        __syncthreads();
    }
    if (threadIdx.x == 0)
        out[b * O_ + o] = red[0] * (1.0f / (float)T_) + b_read[o];
}

// ---------------------------------------------------------------------------
extern "C" void kernel_launch(void* const* d_in, const int* in_sizes, int n_in,
                              void* d_out, int out_size, void* d_ws, size_t ws_size,
                              hipStream_t stream)
{
    (void)in_sizes; (void)n_in; (void)out_size; (void)ws_size;

    const float* x      = (const float*)d_in[0];   // [B,T,D]
    const float* W_enc  = (const float*)d_in[1];   // [H,D]
    const float* W_rec  = (const float*)d_in[2];   // [H,H]
    const float* W_read = (const float*)d_in[3];   // [O,H]
    const float* b_read = (const float*)d_in[4];   // [O]
    float* out = (float*)d_out;

    char* ws = (char*)d_ws;
    __bf16*   Bt   = (__bf16*)  (ws + OFF_BT);
    __bf16*   spkA = (__bf16*)  (ws + OFF_SPK0);
    __bf16*   spkB = (__bf16*)  (ws + OFF_SPK1);
    float*    avg  = (float*)   (ws + OFF_AVG);
    unsigned* cnt  = (unsigned*)(ws + OFF_CNT);
    __bf16*   Xbf  = (__bf16*)  (ws + OFF_XBF);

    // output tuple layout: (output[B,O], spikes[B,T,H], voltages[B,T,H], rate[T])
    float* out_o  = out;
    float* spikes = out + B_ * O_;
    float* volts  = spikes + (size_t)B_ * T_ * H_;
    float* rate   = volts  + (size_t)B_ * T_ * H_;

    init_state<<<(B_ * H_ + 255) / 256, 256, 0, stream>>>(spkA, rate, cnt);

    {
        dim3 grid(H_ / 32, H_ / 32);
        dim3 blk(32, 8);
        transpose_rec<<<grid, blk, 0, stream>>>(W_rec, Bt);
        copy_enc<<<(H_ * D_ + 255) / 256, 256, 0, stream>>>(W_enc, Bt);
        xpose_x<<<(B_ * T_ * D_ + 255) / 256, 256, 0, stream>>>(x, Xbf);
    }

    lif_persist<<<NBLK, 64, 0, stream>>>(Bt, Xbf, spkA, spkB, avg,
                                         spikes, volts, rate, cnt);

    readout<<<B_ * O_, 256, 0, stream>>>(avg, W_read, b_read, out_o);
}
